// WeightedAttGNN_15693810500055
// MI455X (gfx1250) — matrix operand order repaired and checked by reading, not compiled, not run on hardware
//
#include <hip/hip_runtime.h>

typedef __attribute__((ext_vector_type(16))) __bf16 bf16x16;
typedef __attribute__((ext_vector_type(8)))  __bf16 bf16x8;
typedef __attribute__((ext_vector_type(8)))  float  f32x8;

#define DEVINL __device__ __forceinline__

constexpr int N_NODES = 20000;
constexpr int N_EDGES = 320000;
constexpr int N_GRAPH = 64;
constexpr int F_IN    = 1024;
constexpr int HEADS   = 2;
constexpr int CH      = 256;
constexpr int HC      = 512;   // HEADS*CH
constexpr int QKVS    = 2048;  // q|k|v|skip columns
constexpr int OD      = 128;

DEVINL float sigmoidf(float x) { return 1.f / (1.f + __expf(-x)); }
DEVINL float leakyf(float x)   { return x > 0.f ? x : 0.01f * x; }

DEVINL __bf16 f2bf(float f) {
  union { float f; unsigned u; } a; a.f = f;
  unsigned u = (a.u + 0x7fffu + ((a.u >> 16) & 1u)) >> 16;   // RNE
  union { unsigned short s; __bf16 b; } o; o.s = (unsigned short)u;
  return o.b;
}
// monotone float<->uint key for atomicMax-based segment max
DEVINL unsigned fkey(float f) {
  unsigned b = __float_as_uint(f);
  return (b & 0x80000000u) ? ~b : (b | 0x80000000u);
}
DEVINL float funkey(unsigned u) {
  return (u & 0x80000000u) ? __uint_as_float(u & 0x7fffffffu) : __uint_as_float(~u);
}

// ---------------------------------------------------------------------------
// Tiled bf16 WMMA GEMM:  C[M x ldc] (+bias, opt fused leaky+sigmoid residual)
//   A: M x K bf16 row-major        Bt: Nc x K bf16 (B transposed, row-major)
// Block: 256 thr = 8 waves (2 x 4). BM=64 BN=256 BK=32.
// A tile double-buffered in LDS via GLOBAL_LOAD_ASYNC_TO_LDS_B128 (ASYNCcnt):
//   one barrier per K-step: wait asynccnt -> barrier -> issue next -> compute.
// Each wave: 2x4 = 8 v_wmma_f32_16x16x32_bf16 per K-step.
// ---------------------------------------------------------------------------
__global__ __launch_bounds__(256) void wmma_gemm_bf16(
    const __bf16* __restrict__ A, const __bf16* __restrict__ Bt,
    const float* __restrict__ bias, const float* __restrict__ resid,
    float* __restrict__ C, int M, int K, int ldc, int op)
{
  constexpr int BM = 64, BN = 256, LDA = 40;
  __shared__ __align__(16) __bf16 lds_a[2 * BM * LDA];

  const int row0 = blockIdx.x * BM;
  const int col0 = blockIdx.y * BN;
  const int tid  = threadIdx.x;
  const int lane = tid & 31;
  const int wid  = tid >> 5;
  const int half = lane >> 4;      // 0: lanes 0-15, 1: lanes 16-31
  const int lrow = lane & 15;
  const int rbase = (wid & 1) * 32;
  const int cbase = (wid >> 1) * 64;

  f32x8 z = {0.f, 0.f, 0.f, 0.f, 0.f, 0.f, 0.f, 0.f};
  f32x8 acc[2][4] = {{z, z, z, z}, {z, z, z, z}};

  // --- async A staging: each thread moves one 16B chunk per 64x32 tile ---
  const int arow = tid >> 2;                  // 0..63
  const int achk = tid & 3;                   // 0..3 (8 bf16 each)
  long grow = row0 + arow;
  if (grow >= M) grow = M - 1;                // clamp: keeps EXEC all-ones; dup rows discarded
  const __bf16* gsrc = A + grow * (long)K + achk * 8;
  // flat->LDS mapping uses addr[31:0] as the LDS byte address (ISA 10.2)
  unsigned ldst[2];
  ldst[0] = (unsigned)(size_t)&lds_a[arow * LDA + achk * 8];
  ldst[1] = ldst[0] + (unsigned)(BM * LDA * 2);

  auto issue = [&](int buf, int k0) {
    asm volatile("global_load_async_to_lds_b128 %0, %1, off"
                 :: "v"(ldst[buf]), "v"(gsrc + k0)
                 : "memory");
  };

  issue(0, 0);
  const int nk = K >> 5;
  for (int ik = 0; ik < nk; ++ik) {
    const int k0  = ik << 5;
    const int cur = ik & 1;
    asm volatile("s_wait_asynccnt 0" ::: "memory");  // my tile for buf[cur] landed
    __syncthreads();                                  // everyone's landed, prev reads done
    if (ik + 1 < nk) issue(cur ^ 1, k0 + 32);         // overlap next stage with compute

    const __bf16* base = lds_a + cur * (BM * LDA);
    bf16x16 afrag[2], bfrag[4];
#pragma unroll
    for (int t = 0; t < 2; ++t) {
      // A k order used here (contiguous, matches B): lanes0-15 K{0..15}, lanes16-31 K{16..31}
      const __bf16* p = base + (rbase + t * 16 + lrow) * LDA;
      bf16x8 lo = *(const bf16x8*)(p + half * 16);
      bf16x8 hi = *(const bf16x8*)(p + half * 16 + 8);
#pragma unroll
      for (int i = 0; i < 8; ++i) { afrag[t][i] = lo[i]; afrag[t][i + 8] = hi[i]; }
    }
#pragma unroll
    for (int u = 0; u < 4; ++u) {
      const long col = col0 + cbase + u * 16 + lrow;  // B 32x16: lane=N, half selects K 0..15/16..31
      const __bf16* pb = Bt + col * (long)K + k0 + half * 16;
      __builtin_prefetch((const void*)(pb + 32), 0, 3);
      bfrag[u] = *(const bf16x16*)pb;
    }
#pragma unroll
    for (int t = 0; t < 2; ++t)
#pragma unroll
      for (int u = 0; u < 4; ++u)
        acc[t][u] = __builtin_amdgcn_wmma_f32_16x16x32_bf16(
            false, afrag[t], false, bfrag[u], (short)0, acc[t][u], false, false);
  }

  // C layout: VGPR g -> row = 8*half + g, col = lane&15
#pragma unroll
  for (int t = 0; t < 2; ++t) {
    const int rb = row0 + rbase + t * 16 + half * 8;
#pragma unroll
    for (int u = 0; u < 4; ++u) {
      const int c  = col0 + cbase + u * 16 + lrow;
      const float bc = bias ? bias[c] : 0.f;
#pragma unroll
      for (int g = 0; g < 8; ++g) {
        const int r = rb + g;
        if (r < M) {
          float vv = acc[t][u][g] + bc;
          if (op == 1) {                       // fc1 epilogue: leaky + sigmoid(x) residual
            vv = leakyf(vv);
            vv += sigmoidf(resid[(long)r * ldc + c]);
          }
          C[(long)r * ldc + c] = vv;
        }
      }
    }
  }
}

// --------------------------- prep / convert kernels ------------------------
__global__ void prep_qkvs_w(const float* Wq, const float* Wk, const float* Wv,
                            const float* Ws, const float* bq, const float* bk,
                            const float* bv, const float* bs,
                            __bf16* wt, float* bias)
{
  long i = (long)blockIdx.x * 256 + threadIdx.x;   // 2048*1024
  if (i >= (long)QKVS * F_IN) return;
  int j  = (int)(i >> 10);                         // output column 0..2047
  int kk = (int)(i & 1023);
  int sect = j >> 9, jj = j & 511;
  const float* W = sect == 0 ? Wq : sect == 1 ? Wk : sect == 2 ? Wv : Ws;
  wt[(long)j * F_IN + kk] = f2bf(W[(long)kk * HC + jj]);   // transpose
  if (kk == 0) {
    const float* b = sect == 0 ? bq : sect == 1 ? bk : sect == 2 ? bv : bs;
    bias[j] = b[jj];
  }
}

__global__ void prep_fc1_w(const float* W, __bf16* wt) {
  long i = (long)blockIdx.x * 256 + threadIdx.x;   // 1024*512
  if (i >= (long)F_IN * HC) return;
  int o = (int)(i >> 9), kk = (int)(i & 511);
  wt[(long)o * HC + kk] = f2bf(W[(long)kk * F_IN + o]);
}

__global__ void cvt_bf16(const float* __restrict__ x, __bf16* __restrict__ y, long n) {
  long i = (long)blockIdx.x * 256 + threadIdx.x;
  if (i < n) y[i] = f2bf(x[i]);
}

// ------------------------------ attention ----------------------------------
__global__ __launch_bounds__(256) void attn_logits(
    const int* __restrict__ ei, const float* __restrict__ eattr,
    const float* __restrict__ qkvs, const float* __restrict__ We,
    float* __restrict__ alpha, unsigned* __restrict__ mkeys)
{
  long task = (long)blockIdx.x * 8 + (threadIdx.x >> 5);
  int lane = threadIdx.x & 31;
  if (task >= (long)N_EDGES * HEADS) return;
  int e = (int)(task >> 1), h = (int)(task & 1);
  int src = ei[e], dst = ei[N_EDGES + e];
  float ea = eattr[e];
  const float* q = qkvs + (long)dst * QKVS + h * CH;
  const float* k = qkvs + (long)src * QKVS + HC + h * CH;
  const float* w = We + h * CH;
  float s = 0.f;
  for (int c = lane; c < CH; c += 32) s += q[c] * (k[c] + ea * w[c]);
  for (int off = 16; off; off >>= 1) s += __shfl_xor(s, off, 32);
  if (lane == 0) {
    s *= 0.0625f;                                  // 1/sqrt(256)
    alpha[task] = s;
    atomicMax(&mkeys[dst * HEADS + h], fkey(s));
  }
}

__global__ void attn_exp(const int* __restrict__ ei, float* __restrict__ alpha,
                         const unsigned* __restrict__ mkeys, float* __restrict__ denom)
{
  long i = (long)blockIdx.x * 256 + threadIdx.x;
  if (i >= (long)N_EDGES * HEADS) return;
  int e = (int)(i >> 1), h = (int)(i & 1);
  int dst = ei[N_EDGES + e];
  float a = __expf(alpha[i] - funkey(mkeys[dst * HEADS + h]));
  alpha[i] = a;
  atomicAdd(&denom[dst * HEADS + h], a);
}

__global__ __launch_bounds__(256) void attn_scatter(
    const int* __restrict__ ei, const float* __restrict__ eattr,
    const float* __restrict__ We, const float* __restrict__ alpha,
    const float* __restrict__ denom, float* __restrict__ qkvs)
{
  long task = (long)blockIdx.x * 8 + (threadIdx.x >> 5);
  int lane = threadIdx.x & 31;
  if (task >= (long)N_EDGES * HEADS) return;
  int e = (int)(task >> 1), h = (int)(task & 1);
  int src = ei[e], dst = ei[N_EDGES + e];
  float ea = eattr[e];
  float coef = alpha[task] / (denom[dst * HEADS + h] + 1e-16f);
  const float* v = qkvs + (long)src * QKVS + 2 * HC + h * CH;
  const float* w = We + h * CH;
  float* agg = qkvs + (long)dst * QKVS + 3 * HC + h * CH;  // skip section (in place)
  for (int c = lane; c < CH; c += 32)
    atomicAdd(&agg[c], (v[c] + ea * w[c]) * coef);
}

__global__ void postconv(const float* __restrict__ qkvs, __bf16* __restrict__ h1) {
  long i = (long)blockIdx.x * 256 + threadIdx.x;   // N*HC
  if (i >= (long)N_NODES * HC) return;
  int n = (int)(i >> 9), c = (int)(i & 511);
  h1[i] = f2bf(sigmoidf(qkvs[(long)n * QKVS + 3 * HC + c]));
}

// ------------------------------ pooling / tails ----------------------------
__global__ void pool_accum(const float* __restrict__ h2, const int* __restrict__ batch,
                           float* __restrict__ pooled, float* __restrict__ cnt)
{
  long i = (long)blockIdx.x * 256 + threadIdx.x;   // N*F
  if (i >= (long)N_NODES * F_IN) return;
  int n = (int)(i >> 10), c = (int)(i & 1023);
  int b = batch[n];
  atomicAdd(&pooled[(long)b * F_IN + c], h2[i]);
  if (c == 0) atomicAdd(&cnt[b], 1.f);
}

__global__ void branch_fc2(const float* __restrict__ pooled, const float* __restrict__ cnt,
                           const float* __restrict__ W, const float* __restrict__ b,
                           float* __restrict__ out)
{
  int i = blockIdx.x * 256 + threadIdx.x;          // 64*128
  if (i >= N_GRAPH * OD) return;
  int g = i >> 7, o = i & 127;
  float inv = 1.f / fmaxf(cnt[g], 1.f);
  const float* p = pooled + (long)g * F_IN;
  float s = b[o];
  for (int k = 0; k < F_IN; ++k) s += (p[k] * inv) * W[(long)k * OD + o];
  out[i] = leakyf(s);
}

__global__ __launch_bounds__(256) void head_kernel(
    const float* __restrict__ x1, const float* __restrict__ x2,
    const float* __restrict__ W1, const float* __restrict__ b1,
    const float* __restrict__ W2, const float* __restrict__ b2,
    const float* __restrict__ Wo, const float* __restrict__ bo,
    float* __restrict__ out)
{
  __shared__ float xc[256], t1[256], t2[64];
  int g = blockIdx.x, j = threadIdx.x;
  xc[j] = (j < OD) ? x1[g * OD + j] : x2[g * OD + (j - OD)];
  __syncthreads();
  float s = b1[j];
  for (int k = 0; k < 256; ++k) s += xc[k] * W1[k * 256 + j];
  t1[j] = leakyf(s);
  __syncthreads();
  if (j < 64) {
    float s2 = b2[j];
    for (int k = 0; k < 256; ++k) s2 += t1[k] * W2[k * 64 + j];
    t2[j] = leakyf(s2);
  }
  __syncthreads();
  if (j == 0) {
    float s3 = bo[0];
    for (int k = 0; k < 64; ++k) s3 += t2[k] * Wo[k];
    out[g] = sigmoidf(s3);
  }
}

// ------------------------------- driver ------------------------------------
extern "C" void kernel_launch(void* const* d_in, const int* in_sizes, int n_in,
                              void* d_out, int out_size, void* d_ws, size_t ws_size,
                              hipStream_t stream)
{
  (void)in_sizes; (void)n_in; (void)out_size; (void)ws_size;
  char* ws = (char*)d_ws;
  size_t off = 0;
  auto alloc = [&](size_t bytes) -> void* {
    void* p = ws + off;
    off = (off + bytes + 255) & ~(size_t)255;
    return p;
  };

  float*    bo1       = (float*)   alloc((size_t)N_GRAPH * OD * 4);
  float*    bo2       = (float*)   alloc((size_t)N_GRAPH * OD * 4);
  // --- per-branch scratch (reused between branches) ---
  __bf16*   wt_qkvs   = (__bf16*)  alloc((size_t)QKVS * F_IN * 2);
  float*    bias_qkvs = (float*)   alloc((size_t)QKVS * 4);
  __bf16*   wt_fc1    = (__bf16*)  alloc((size_t)F_IN * HC * 2);
  __bf16*   x_bf      = (__bf16*)  alloc((size_t)N_NODES * F_IN * 2);
  float*    alpha     = (float*)   alloc((size_t)N_EDGES * HEADS * 4);
  unsigned* mkeys     = (unsigned*)alloc((size_t)N_NODES * HEADS * 4);
  float*    denom     = (float*)   alloc((size_t)N_NODES * HEADS * 4);
  __bf16*   h1_bf     = (__bf16*)  alloc((size_t)N_NODES * HC * 2);
  float*    pooled    = (float*)   alloc((size_t)N_GRAPH * F_IN * 4);
  float*    cnt       = (float*)   alloc((size_t)N_GRAPH * 4);
  float*    qkvs      = (float*)   alloc((size_t)N_NODES * QKVS * 4);
  float*    h2        = qkvs;   // overlay: qkvs is dead once h1_bf is built

  auto run_branch = [&](int xi, int pbase, float* bout) {
    const float* x   = (const float*)d_in[xi];
    const int*   ei  = (const int*)  d_in[xi + 1];
    const float* ea  = (const float*)d_in[xi + 2];
    const int*   bat = (const int*)  d_in[xi + 3];
    const float* Wq  = (const float*)d_in[pbase + 0];
    const float* bq  = (const float*)d_in[pbase + 1];
    const float* Wk  = (const float*)d_in[pbase + 2];
    const float* bk  = (const float*)d_in[pbase + 3];
    const float* Wv  = (const float*)d_in[pbase + 4];
    const float* bv  = (const float*)d_in[pbase + 5];
    const float* We  = (const float*)d_in[pbase + 6];
    const float* Ws  = (const float*)d_in[pbase + 7];
    const float* bs  = (const float*)d_in[pbase + 8];
    const float* f1W = (const float*)d_in[pbase + 9];
    const float* f1b = (const float*)d_in[pbase + 10];
    const float* f2W = (const float*)d_in[pbase + 11];
    const float* f2b = (const float*)d_in[pbase + 12];

    prep_qkvs_w<<<(QKVS * F_IN) / 256, 256, 0, stream>>>(Wq, Wk, Wv, Ws, bq, bk, bv, bs,
                                                         wt_qkvs, bias_qkvs);
    prep_fc1_w<<<(F_IN * HC) / 256, 256, 0, stream>>>(f1W, wt_fc1);
    cvt_bf16<<<(N_NODES * F_IN) / 256, 256, 0, stream>>>(x, x_bf, (long)N_NODES * F_IN);
    hipMemsetAsync(mkeys, 0, (size_t)N_NODES * HEADS * 4, stream);
    hipMemsetAsync(denom, 0, (size_t)N_NODES * HEADS * 4, stream);

    dim3 g1((N_NODES + 63) / 64, QKVS / 256);
    wmma_gemm_bf16<<<g1, 256, 0, stream>>>(x_bf, wt_qkvs, bias_qkvs, nullptr, qkvs,
                                           N_NODES, F_IN, QKVS, 0);

    const int nt = N_EDGES * HEADS;
    attn_logits<<<(nt + 7) / 8, 256, 0, stream>>>(ei, ea, qkvs, We, alpha, mkeys);
    attn_exp<<<(nt + 255) / 256, 256, 0, stream>>>(ei, alpha, mkeys, denom);
    attn_scatter<<<(nt + 7) / 8, 256, 0, stream>>>(ei, ea, We, alpha, denom, qkvs);
    postconv<<<(N_NODES * HC) / 256, 256, 0, stream>>>(qkvs, h1_bf);

    dim3 g2((N_NODES + 63) / 64, F_IN / 256);
    wmma_gemm_bf16<<<g2, 256, 0, stream>>>(h1_bf, wt_fc1, f1b, x, h2,
                                           N_NODES, HC, F_IN, 1);

    hipMemsetAsync(pooled, 0, (size_t)N_GRAPH * F_IN * 4, stream);
    hipMemsetAsync(cnt, 0, (size_t)N_GRAPH * 4, stream);
    pool_accum<<<(N_NODES * F_IN) / 256, 256, 0, stream>>>(h2, bat, pooled, cnt);
    branch_fc2<<<(N_GRAPH * OD) / 256, 256, 0, stream>>>(pooled, cnt, f2W, f2b, bout);
  };

  run_branch(0, 8,  bo1);   // protein 1
  run_branch(4, 21, bo2);   // protein 2

  head_kernel<<<N_GRAPH, 256, 0, stream>>>(
      bo1, bo2,
      (const float*)d_in[34], (const float*)d_in[35],
      (const float*)d_in[36], (const float*)d_in[37],
      (const float*)d_in[38], (const float*)d_in[39],
      (float*)d_out);
}
